// PosteriorRotationLayer_446676599068
// MI455X (gfx1250) — matrix operand order repaired and checked by reading, not compiled, not run on hardware
//
#include <hip/hip_runtime.h>

#define H_      16
#define D_      1024
#define dH      64
#define T_      2048
#define TC_     2016
#define NT_     (TC_ / 16)       // 126 n-tiles in the skew GEMM
#define EPS_ROT 0.01f
#define LN_EPS  1e-5f

typedef __attribute__((ext_vector_type(16))) __bf16        v16bf;
typedef __attribute__((ext_vector_type(8)))  float         v8f;
typedef __attribute__((ext_vector_type(4)))  unsigned int  v4u;

union BFrag {
    v16bf          v;
    unsigned short u[16];
    v4u            q[2];
};

__device__ __forceinline__ unsigned short f2bf_bits(float f) {
    unsigned u = __builtin_bit_cast(unsigned, f);
    // round-to-nearest-even bf16
    return (unsigned short)((u + 0x7FFFu + ((u >> 16) & 1u)) >> 16);
}

// ---------------------------------------------------------------------------
// Prep: elementwise f32 -> bf16
// ---------------------------------------------------------------------------
__global__ void cvt_bf16_kernel(const float* __restrict__ src,
                                unsigned short* __restrict__ dst, int n) {
    const int i = blockIdx.x * blockDim.x + threadIdx.x;
    if (i < n) dst[i] = f2bf_bits(src[i]);
}

// Prep: f32 (rows x cols) -> bf16 transposed (cols x rows), i.e. (N,K) layout
__global__ void cvt_bf16_t_kernel(const float* __restrict__ src,
                                  unsigned short* __restrict__ dst,
                                  int rows, int cols) {
    const int i = blockIdx.x * blockDim.x + threadIdx.x;
    if (i < rows * cols) {
        const int r = i / cols;
        const int c = i - r * cols;
        dst[(size_t)c * rows + r] = f2bf_bits(src[i]);
    }
}

// ---------------------------------------------------------------------------
// q = xh @ Wq + bq  per head  ->  qbf (H, T, 64) bf16
// grid (T/16, H), block 128 (4 waves, wave w owns 16-col subtile w)
// WqT is (N=64, K=64) row-major bf16: B fragment loads are contiguous.
// ---------------------------------------------------------------------------
__global__ void qproj_kernel(const unsigned short* __restrict__ xbf,
                             const unsigned short* __restrict__ WqT,
                             const float* __restrict__ bq,
                             unsigned short* __restrict__ qbf) {
    const int tile = blockIdx.x;
    const int h    = blockIdx.y;
    const int tid  = threadIdx.x;
    const int wave = tid >> 5;
    const int lane = tid & 31;
    const int half = lane >> 4;
    const int lo   = lane & 15;
    const int t0   = tile * 16;
    const int col  = wave * 16 + lo;

    const unsigned short* arow = xbf + (size_t)(t0 + lo) * D_ + h * dH;
    const unsigned short* brow = WqT + (size_t)col * dH;

    v8f acc = {};
#pragma unroll
    for (int c = 0; c < 2; ++c) {
        const int kb = c * 32;
        BFrag a, b;
        a.q[0] = *(const v4u*)(arow + kb + 8 * half);
        a.q[1] = *(const v4u*)(arow + kb + 16 + 8 * half);
        b.q[0] = *(const v4u*)(brow + kb + 8 * half);
        b.q[1] = *(const v4u*)(brow + kb + 16 + 8 * half);
        acc = __builtin_amdgcn_wmma_f32_16x16x32_bf16(false, a.v, false, b.v,
                                                      (short)0, acc, false, false);
    }
    const float bias = bq[col];
#pragma unroll
    for (int r = 0; r < 8; ++r) {
        const int m = r + 8 * half;
        qbf[((size_t)h * T_ + (t0 + m)) * dH + col] = f2bf_bits(acc[r] + bias);
    }
}

// ---------------------------------------------------------------------------
// kv = context @ Wkv + bkv, then per-(s,head) layernorm (torch-style,
// unbiased std) -> kvn (H, Tc, 64) bf16
// grid (Tc/16, H), block 128 (4 waves, wave w owns 16-col subtile w)
// WkvT is (N=1024, K=1024) row-major bf16.
// ---------------------------------------------------------------------------
__global__ void kvln_kernel(const unsigned short* __restrict__ ctxbf,
                            const unsigned short* __restrict__ WkvT,
                            const float* __restrict__ bkv,
                            const float* __restrict__ gamma,
                            const float* __restrict__ beta,
                            unsigned short* __restrict__ kvn) {
    __shared__ float kvt[16 * dH];
    __shared__ float mu[16];
    __shared__ float rinv[16];

    const int stile = blockIdx.x;
    const int h     = blockIdx.y;
    const int tid   = threadIdx.x;
    const int wave  = tid >> 5;
    const int lane  = tid & 31;
    const int half  = lane >> 4;
    const int lo    = lane & 15;
    const int s0    = stile * 16;
    const int lcol  = wave * 16 + lo;     // col within head slice
    const int col   = h * dH + lcol;      // col within D

    const unsigned short* arow = ctxbf + (size_t)(s0 + lo) * D_;
    const unsigned short* brow = WkvT + (size_t)col * D_;

    v8f acc = {};
#pragma unroll 4
    for (int c = 0; c < 32; ++c) {
        const int kb = c * 32;
        BFrag a, b;
        a.q[0] = *(const v4u*)(arow + kb + 8 * half);
        a.q[1] = *(const v4u*)(arow + kb + 16 + 8 * half);
        b.q[0] = *(const v4u*)(brow + kb + 8 * half);
        b.q[1] = *(const v4u*)(brow + kb + 16 + 8 * half);
        acc = __builtin_amdgcn_wmma_f32_16x16x32_bf16(false, a.v, false, b.v,
                                                      (short)0, acc, false, false);
    }
    const float bias = bkv[col];
#pragma unroll
    for (int r = 0; r < 8; ++r)
        kvt[(r + 8 * half) * dH + lcol] = acc[r] + bias;
    __syncthreads();

    if (tid < 16) {
        float s = 0.f, ss = 0.f;
#pragma unroll 8
        for (int k = 0; k < dH; ++k) {
            const float v = kvt[tid * dH + k];
            s  += v;
            ss += v * v;
        }
        const float mean = s * (1.0f / dH);
        const float var  = fmaxf((ss - dH * mean * mean) * (1.0f / (dH - 1)), 0.0f);
        mu[tid]   = mean;
        rinv[tid] = 1.0f / (sqrtf(var) + LN_EPS);
    }
    __syncthreads();

#pragma unroll
    for (int e = 0; e < 8; ++e) {
        const int idx = tid * 8 + e;
        const int row = idx >> 6;
        const int k   = idx & 63;
        float v = (kvt[idx] - mu[row]) * rinv[row];
        v = gamma[k] * v + beta[k];
        kvn[((size_t)h * TC_ + (s0 + row)) * dH + k] = f2bf_bits(v);
    }
}

// ---------------------------------------------------------------------------
// Fused skew-GEMM + tanh + skew-symmetric rotation apply.
// grid (T/16, H), block 128. Per block: 16-token tile of one head.
//   S[t,s] = tanh(q[t,:].kvn[s,:]),  s = j(j-1)/2 + i  (j>i)
//   delta[t,j] += xh[t,i]*S ; delta[t,i] -= xh[t,j]*S   (LDS ds_add_f32)
//   out = xh + eps*delta
// Column lo of n-tile nt handles s = lo*126 + nt: spreading s across the
// triangle makes the 16 decoded j values distinct -> conflict-free atomics.
// ---------------------------------------------------------------------------
__global__ void rot_kernel(const float* __restrict__ x,
                           const unsigned short* __restrict__ qbf,
                           const unsigned short* __restrict__ kvn,
                           float* __restrict__ out) {
    __shared__ float xh_lds[16 * dH];
    __shared__ float delta_lds[16 * dH];

    const int tile = blockIdx.x;
    const int h    = blockIdx.y;
    const int tid  = threadIdx.x;
    const int wave = tid >> 5;
    const int lane = tid & 31;
    const int half = lane >> 4;
    const int lo   = lane & 15;
    const int t0   = tile * 16;

    // stage xh tile (f32) and clear delta
#pragma unroll
    for (int e = 0; e < 8; ++e) {
        const int idx = tid * 8 + e;
        const int row = idx >> 6;
        const int k   = idx & 63;
        xh_lds[idx]    = x[(size_t)(t0 + row) * D_ + h * dH + k];
        delta_lds[idx] = 0.0f;
    }
    __syncthreads();

    // loop-invariant A fragments (q tile, K=64 -> two 16x16x32 chunks)
    BFrag a0, a1;
    const unsigned short* qrow = qbf + ((size_t)h * T_ + (t0 + lo)) * dH;
    a0.q[0] = *(const v4u*)(qrow + 8 * half);
    a0.q[1] = *(const v4u*)(qrow + 16 + 8 * half);
    a1.q[0] = *(const v4u*)(qrow + 32 + 8 * half);
    a1.q[1] = *(const v4u*)(qrow + 48 + 8 * half);

    const unsigned short* kvh = kvn + (size_t)h * TC_ * dH;

    for (int nt = wave; nt < NT_; nt += 4) {
        const int s = lo * NT_ + nt;              // interleaved column s
        const unsigned short* krow = kvh + (size_t)s * dH;
        BFrag b0, b1;
        b0.q[0] = *(const v4u*)(krow + 8 * half);
        b0.q[1] = *(const v4u*)(krow + 16 + 8 * half);
        b1.q[0] = *(const v4u*)(krow + 32 + 8 * half);
        b1.q[1] = *(const v4u*)(krow + 48 + 8 * half);
        if (nt + 4 < NT_)
            __builtin_prefetch(krow + (size_t)4 * dH, 0, 1);

        v8f acc = {};
        acc = __builtin_amdgcn_wmma_f32_16x16x32_bf16(false, a0.v, false, b0.v,
                                                      (short)0, acc, false, false);
        acc = __builtin_amdgcn_wmma_f32_16x16x32_bf16(false, a1.v, false, b1.v,
                                                      (short)0, acc, false, false);

        // decode s -> (j,i), j>i, s = j(j-1)/2 + i  (once per tile per lane)
        int j = (int)((1.0f + sqrtf(1.0f + 8.0f * (float)s)) * 0.5f);
        while (j * (j - 1) / 2 > s) --j;
        while ((j + 1) * j / 2 <= s) ++j;
        const int i = s - j * (j - 1) / 2;

#pragma unroll
        for (int r = 0; r < 8; ++r) {
            const int row = r + 8 * half;         // token within tile
            float xx = fminf(fmaxf(acc[r], -15.0f), 15.0f);
            const float e2 = __expf(2.0f * xx);
            const float sv = 1.0f - 2.0f / (e2 + 1.0f);   // tanh
            const float xi = xh_lds[row * dH + i];
            const float xj = xh_lds[row * dH + j];
            atomicAdd(&delta_lds[row * dH + j],  xi * sv);
            atomicAdd(&delta_lds[row * dH + i], -xj * sv);
        }
    }
    __syncthreads();

#pragma unroll
    for (int e = 0; e < 8; ++e) {
        const int idx = tid * 8 + e;
        const int row = idx >> 6;
        const int k   = idx & 63;
        out[(size_t)(t0 + row) * D_ + h * dH + k] =
            xh_lds[idx] + EPS_ROT * delta_lds[idx];
    }
}

// ---------------------------------------------------------------------------
extern "C" void kernel_launch(void* const* d_in, const int* in_sizes, int n_in,
                              void* d_out, int out_size, void* d_ws, size_t ws_size,
                              hipStream_t stream) {
    (void)in_sizes; (void)n_in; (void)out_size; (void)ws_size;

    const float* x     = (const float*)d_in[0];
    const float* ctx   = (const float*)d_in[1];
    const float* Wq    = (const float*)d_in[2];
    const float* bq    = (const float*)d_in[3];
    const float* Wkv   = (const float*)d_in[4];
    const float* bkv   = (const float*)d_in[5];
    const float* gamma = (const float*)d_in[6];
    const float* beta  = (const float*)d_in[7];
    float* out = (float*)d_out;

    // workspace layout (all bf16 stored as ushort; every block 16B aligned)
    unsigned short* qbf   = (unsigned short*)d_ws;                  // H*T*64
    unsigned short* kvn   = qbf   + (size_t)H_ * T_  * dH;          // H*Tc*64
    unsigned short* xbf   = kvn   + (size_t)H_ * TC_ * dH;          // T*D
    unsigned short* ctxbf = xbf   + (size_t)T_ * D_;                // Tc*D
    unsigned short* WkvT  = ctxbf + (size_t)TC_ * D_;               // D*D (N,K)
    unsigned short* WqT   = WkvT  + (size_t)D_ * D_;                // d*d (N,K)

    const int nx = T_ * D_;          // 2,097,152
    const int nc = TC_ * D_;         // 2,064,384
    cvt_bf16_kernel  <<<(nx + 255) / 256, 256, 0, stream>>>(x,   xbf,   nx);
    cvt_bf16_kernel  <<<(nc + 255) / 256, 256, 0, stream>>>(ctx, ctxbf, nc);
    cvt_bf16_t_kernel<<<(D_ * D_ + 255) / 256, 256, 0, stream>>>(Wkv, WkvT, D_, D_);
    cvt_bf16_t_kernel<<<(dH * dH + 255) / 256, 256, 0, stream>>>(Wq,  WqT,  dH, dH);

    qproj_kernel<<<dim3(T_ / 16, H_),  128, 0, stream>>>(xbf, WqT, bq, qbf);
    kvln_kernel <<<dim3(TC_ / 16, H_), 128, 0, stream>>>(ctxbf, WkvT, bkv, gamma, beta, kvn);
    rot_kernel  <<<dim3(T_ / 16, H_),  128, 0, stream>>>(x, qbf, kvn, out);
}
